// PUNet_63144609186367
// MI455X (gfx1250) — matrix operand order repaired and checked by reading, not compiled, not run on hardware
//
#include <hip/hip_runtime.h>

typedef __attribute__((ext_vector_type(16))) _Float16 v16h;
typedef __attribute__((ext_vector_type(8)))  _Float16 v8h;
typedef __attribute__((ext_vector_type(8)))  float    v8f;

#define NB      16
#define NPTS    4096
#define NPOINT  1024
#define NSAMP   32
#define CIN     64
#define KPAD    96
#define R2      0.04f

// ---------------------------------------------------------------------------
// Kernel 0: convert weights to f16, pad layer-1 K to 96 with bias trick
//   w1h[n][c] : c<67 -> W1[n][c], c==67 -> b1[n] (x[67]==1), else 0
//   w2h = W2 (128x128), w3h = W3 (256x128), all row-major [out][in]
// ---------------------------------------------------------------------------
__global__ __launch_bounds__(256) void prep_kernel(
    const float* __restrict__ W1, const float* __restrict__ b1,
    const float* __restrict__ W2, const float* __restrict__ W3,
    _Float16* __restrict__ w1h, _Float16* __restrict__ w2h,
    _Float16* __restrict__ w3h) {
  const int t = blockIdx.x * 256 + threadIdx.x;
  if (t < 128 * KPAD) {
    const int n = t / KPAD, c = t - n * KPAD;
    float v = (c < 67) ? W1[n * 67 + c] : ((c == 67) ? b1[n] : 0.0f);
    w1h[t] = (_Float16)v;
  }
  if (t < 128 * 128) w2h[t] = (_Float16)W2[t];
  if (t < 256 * 128) w3h[t] = (_Float16)W3[t];
}

// ---------------------------------------------------------------------------
// Kernel 1: farthest point sampling. One block per batch, xyz in LDS,
// per-thread distance registers (4 points/thread), shuffle+LDS argmax.
// ---------------------------------------------------------------------------
__global__ __launch_bounds__(1024) void fps_kernel(const float* __restrict__ xyz,
                                                   float* __restrict__ newxyz) {
  __shared__ float px[NPTS], py[NPTS], pz[NPTS];
  __shared__ float wbest[32];
  __shared__ int   warg[32];
  __shared__ int   s_far;
  const int b = blockIdx.x;
  const int t = threadIdx.x;
  const float* base = xyz + (size_t)b * NPTS * 3;
  for (int i = t; i < NPTS; i += 1024) {
    px[i] = base[3 * i + 0];
    py[i] = base[3 * i + 1];
    pz[i] = base[3 * i + 2];
  }
  float dreg[4];
#pragma unroll
  for (int j = 0; j < 4; ++j) dreg[j] = 1e10f;
  if (t == 0) s_far = 0;
  __syncthreads();

  for (int it = 0; it < NPOINT; ++it) {
    const int far = s_far;
    const float cx = px[far], cy = py[far], cz = pz[far];
    if (t == 0) {
      float* o = newxyz + ((size_t)b * NPOINT + it) * 3;
      o[0] = cx; o[1] = cy; o[2] = cz;
    }
    float best = -1.0f;
    int   bi   = 0;
#pragma unroll
    for (int j = 0; j < 4; ++j) {
      const int i = t + j * 1024;
      const float dx = px[i] - cx, dy = py[i] - cy, dz = pz[i] - cz;
      const float d = dx * dx + dy * dy + dz * dz;
      dreg[j] = fminf(dreg[j], d);
      if (dreg[j] > best) { best = dreg[j]; bi = i; }
    }
#pragma unroll
    for (int off = 16; off > 0; off >>= 1) {
      const float ob = __shfl_xor(best, off, 32);
      const int   oi = __shfl_xor(bi, off, 32);
      if (ob > best || (ob == best && oi < bi)) { best = ob; bi = oi; }
    }
    if ((t & 31) == 0) { wbest[t >> 5] = best; warg[t >> 5] = bi; }
    __syncthreads();
    if (t < 32) {
      best = wbest[t];
      bi   = warg[t];
#pragma unroll
      for (int off = 16; off > 0; off >>= 1) {
        const float ob = __shfl_xor(best, off, 32);
        const int   oi = __shfl_xor(bi, off, 32);
        if (ob > best || (ob == best && oi < bi)) { best = ob; bi = oi; }
      }
      if (t == 0) s_far = bi;
    }
    __syncthreads();
  }
}

// ---------------------------------------------------------------------------
// Kernel 2: fused ball-query + gather + 3x WMMA MLP + max-pool.
// 256 threads (8 waves), 2 query groups (64 rows) per block.
// Wave w: M-tile mt = w>>1 (16 rows), N-half nh = w&1.
// WMMA fragment packing matches CDNA5 ISA 16-bit A(16x32)/B(32x16)/f32 C.
// ---------------------------------------------------------------------------
__global__ __launch_bounds__(256) void punet_fused_kernel(
    const float* __restrict__ xyz, const float* __restrict__ feats,
    const float* __restrict__ newxyz,
    const _Float16* __restrict__ w1h, const _Float16* __restrict__ w2h,
    const _Float16* __restrict__ w3h,
    const float* __restrict__ b2v, const float* __restrict__ b3v,
    float* __restrict__ out) {
  __shared__ __align__(16) _Float16 sx[64 * KPAD];    // 12 KB input tile
  __shared__ __align__(16) _Float16 sh1[64 * 128];    // 16 KB layer1 out
  __shared__ __align__(16) _Float16 sh2[64 * 128];    // 16 KB layer2 out
  __shared__ float swmax[4 * 256];                    // per-Mtile col max
  __shared__ int   sel[2 * NSAMP];

  const int lane = threadIdx.x & 31;
  const int wave = threadIdx.x >> 5;
  const int bb   = (int)((blockIdx.x * 2) >> 10);     // batch index
  const float* pb = xyz + (size_t)bb * NPTS * 3;

  // ---- phase A: ball query (waves 0,1, one per group) ----
  if (wave < 2) {
    const int gi = blockIdx.x * 2 + wave;
    const float qx = newxyz[gi * 3 + 0];
    const float qy = newxyz[gi * 3 + 1];
    const float qz = newxyz[gi * 3 + 2];
    int cnt = 0;
    for (int c0 = 0; c0 < NPTS && cnt < NSAMP; c0 += 32) {
      const int p = c0 + lane;
      const float dx = pb[3 * p + 0] - qx;
      const float dy = pb[3 * p + 1] - qy;
      const float dz = pb[3 * p + 2] - qz;
      const bool in = (dx * dx + dy * dy + dz * dz) <= R2;
      const unsigned m = __builtin_amdgcn_ballot_w32(in);
      const int slot = cnt + __popc(m & ((1u << lane) - 1u));
      if (in && slot < NSAMP) sel[wave * NSAMP + slot] = p;
      cnt += __popc(m);
    }
    if (cnt < NSAMP) {
      int f = 0;
      if (cnt > 0) f = sel[wave * NSAMP];
      for (int k = cnt + lane; k < NSAMP; k += 32) sel[wave * NSAMP + k] = f;
    }
  }
  __syncthreads();

  // ---- phase B: gather 64 x 96 f16 tile (xyz-rel | feats | 1 | 0-pad) ----
  const float* fbp = feats + (size_t)bb * CIN * NPTS;
  for (int e = threadIdx.x; e < 64 * KPAD; e += 256) {
    const int row = e / KPAD, c = e - row * KPAD;
    const int g = row >> 5, k = row & 31;
    const int p = sel[g * NSAMP + k];
    float v;
    if (c < 3)        v = pb[3 * p + c] - newxyz[(blockIdx.x * 2 + g) * 3 + c];
    else if (c < 67)  v = fbp[(size_t)(c - 3) * NPTS + p];
    else if (c == 67) v = 1.0f;   // bias column for layer 1
    else              v = 0.0f;
    sx[row * KPAD + c] = (_Float16)v;
  }
  __syncthreads();

  const int mt    = wave >> 1;
  const int nh    = wave & 1;
  const int mrow  = mt * 16 + (lane & 15);
  const int koff  = (lane >> 4) * 8;   // A: lanes 16-31 hold K+8 / K+24
  const int kb    = (lane >> 4) * 16;  // unused; B loads use koff, lanes 16-31 hold K+8 / K+24
  const int rbase = (lane >> 4) * 8;   // C: lanes 16-31 hold rows M+8
  const int ncol  = lane & 15;

  // ---- layer 1: (64x96) x (96x128), bias folded into column 67 ----
  {
    v16h a[3];
#pragma unroll
    for (int kk = 0; kk < 3; ++kk) {
      const v8h lo = *(const v8h*)(sx + mrow * KPAD + kk * 32 + koff);
      const v8h hi = *(const v8h*)(sx + mrow * KPAD + kk * 32 + koff + 16);
#pragma unroll
      for (int i = 0; i < 8; ++i) { a[kk][i] = lo[i]; a[kk][i + 8] = hi[i]; }
    }
#pragma unroll
    for (int ntl = 0; ntl < 4; ++ntl) {
      const int n = (nh * 4 + ntl) * 16 + ncol;
      v8f acc = {};
#pragma unroll
      for (int kk = 0; kk < 3; ++kk) {
        const _Float16* wp = w1h + n * KPAD + kk * 32 + koff;
        const v8h lo = *(const v8h*)wp;
        const v8h hi = *(const v8h*)(wp + 16);
        v16h bm;
#pragma unroll
        for (int i = 0; i < 8; ++i) { bm[i] = lo[i]; bm[i + 8] = hi[i]; }
        acc = __builtin_amdgcn_wmma_f32_16x16x32_f16(false, a[kk], false, bm,
                                                     (short)0, acc, false, false);
      }
#pragma unroll
      for (int r = 0; r < 8; ++r) {
        float v = acc[r];
        v = v > 0.0f ? v : 0.0f;
        sh1[(mt * 16 + rbase + r) * 128 + n] = (_Float16)v;
      }
    }
  }
  __syncthreads();

  // ---- layer 2: (64x128) x (128x128) + b2, relu ----
  {
    v16h a[4];
#pragma unroll
    for (int kk = 0; kk < 4; ++kk) {
      const v8h lo = *(const v8h*)(sh1 + mrow * 128 + kk * 32 + koff);
      const v8h hi = *(const v8h*)(sh1 + mrow * 128 + kk * 32 + koff + 16);
#pragma unroll
      for (int i = 0; i < 8; ++i) { a[kk][i] = lo[i]; a[kk][i + 8] = hi[i]; }
    }
#pragma unroll
    for (int ntl = 0; ntl < 4; ++ntl) {
      const int n = (nh * 4 + ntl) * 16 + ncol;
      v8f acc = {};
#pragma unroll
      for (int kk = 0; kk < 4; ++kk) {
        const _Float16* wp = w2h + n * 128 + kk * 32 + koff;
        const v8h lo = *(const v8h*)wp;
        const v8h hi = *(const v8h*)(wp + 16);
        v16h bm;
#pragma unroll
        for (int i = 0; i < 8; ++i) { bm[i] = lo[i]; bm[i + 8] = hi[i]; }
        acc = __builtin_amdgcn_wmma_f32_16x16x32_f16(false, a[kk], false, bm,
                                                     (short)0, acc, false, false);
      }
      const float bias = b2v[n];
#pragma unroll
      for (int r = 0; r < 8; ++r) {
        float v = acc[r] + bias;
        v = v > 0.0f ? v : 0.0f;
        sh2[(mt * 16 + rbase + r) * 128 + n] = (_Float16)v;
      }
    }
  }
  __syncthreads();

  // ---- layer 3: (64x128) x (128x256) + b3, relu, in-register max-pool ----
  {
    v16h a[4];
#pragma unroll
    for (int kk = 0; kk < 4; ++kk) {
      const v8h lo = *(const v8h*)(sh2 + mrow * 128 + kk * 32 + koff);
      const v8h hi = *(const v8h*)(sh2 + mrow * 128 + kk * 32 + koff + 16);
#pragma unroll
      for (int i = 0; i < 8; ++i) { a[kk][i] = lo[i]; a[kk][i + 8] = hi[i]; }
    }
#pragma unroll
    for (int ntl = 0; ntl < 8; ++ntl) {
      const int n = (nh * 8 + ntl) * 16 + ncol;
      v8f acc = {};
#pragma unroll
      for (int kk = 0; kk < 4; ++kk) {
        const _Float16* wp = w3h + n * 128 + kk * 32 + koff;
        const v8h lo = *(const v8h*)wp;
        const v8h hi = *(const v8h*)(wp + 16);
        v16h bm;
#pragma unroll
        for (int i = 0; i < 8; ++i) { bm[i] = lo[i]; bm[i + 8] = hi[i]; }
        acc = __builtin_amdgcn_wmma_f32_16x16x32_f16(false, a[kk], false, bm,
                                                     (short)0, acc, false, false);
      }
      const float bias = b3v[n];
      float m = 0.0f;  // relu outputs are >= 0
#pragma unroll
      for (int r = 0; r < 8; ++r) {
        float v = acc[r] + bias;
        v = v > 0.0f ? v : 0.0f;
        m = fmaxf(m, v);
      }
      m = fmaxf(m, __shfl_xor(m, 16, 32));  // combine rows 0-7 with 8-15
      if (lane < 16) swmax[mt * 256 + n] = m;
    }
  }
  __syncthreads();

  // ---- final: combine two M-tiles per group, write (B,256,S) ----
  for (int e = threadIdx.x; e < 2 * 256; e += 256) {
    const int g = e >> 8, n = e & 255;
    const float v = fmaxf(swmax[(2 * g) * 256 + n], swmax[(2 * g + 1) * 256 + n]);
    const int gi = blockIdx.x * 2 + g;
    const int s  = gi & 1023;
    out[((size_t)bb * 256 + n) * 1024 + s] = v;
  }
}

// ---------------------------------------------------------------------------
extern "C" void kernel_launch(void* const* d_in, const int* in_sizes, int n_in,
                              void* d_out, int out_size, void* d_ws, size_t ws_size,
                              hipStream_t stream) {
  (void)in_sizes; (void)n_in; (void)out_size; (void)ws_size;
  const float* xyz   = (const float*)d_in[0];
  const float* feats = (const float*)d_in[1];
  const float* W1    = (const float*)d_in[2];
  const float* b1    = (const float*)d_in[3];
  const float* W2    = (const float*)d_in[4];
  const float* b2    = (const float*)d_in[5];
  const float* W3    = (const float*)d_in[6];
  const float* b3    = (const float*)d_in[7];
  float* out = (float*)d_out;

  char* ws = (char*)d_ws;
  float*    newxyz = (float*)(ws + 0);          // 16*1024*3 f32 = 196608 B
  _Float16* w1h    = (_Float16*)(ws + 200704);  // 128*96  f16 = 24576 B
  _Float16* w2h    = (_Float16*)(ws + 225280);  // 128*128 f16 = 32768 B
  _Float16* w3h    = (_Float16*)(ws + 258048);  // 256*128 f16 = 65536 B

  prep_kernel<<<128, 256, 0, stream>>>(W1, b1, W2, W3, w1h, w2h, w3h);
  fps_kernel<<<NB, 1024, 0, stream>>>(xyz, newxyz);
  punet_fused_kernel<<<NB * NPOINT / 2, 256, 0, stream>>>(
      xyz, feats, newxyz, w1h, w2h, w3h, b2, b3, out);
}
